// GNNClassifier_16097537425459
// MI455X (gfx1250) — compile-verified, matrix operand order checked by reading, not run on hardware
//
#include <hip/hip_runtime.h>
#include <hip/hip_bf16.h>
#include <math.h>

typedef __attribute__((ext_vector_type(16))) _Float16 v16h;
typedef __attribute__((ext_vector_type(8)))  float    v8f;

#define N_NODES   100000
#define N_EDGES   1600000
#define N_FEAT    128
#define HIDDEN    64
#define N_CLASSES 10
#define N_GRAPHS  64

// ---------------- degree / norm ----------------
__global__ void k_init_deg(float* deg, int n) {
    int i = blockIdx.x * blockDim.x + threadIdx.x;
    if (i < n) deg[i] = 1.0f;                 // self-loop contributes 1
}

__global__ void k_count_deg(const long long* __restrict__ dst, float* deg, int e) {
    int i = blockIdx.x * blockDim.x + threadIdx.x;
    if (i < e) unsafeAtomicAdd(&deg[(int)dst[i]], 1.0f);
}

__global__ void k_finalize_dinv(float* d, int n) {
    int i = blockIdx.x * blockDim.x + threadIdx.x;
    if (i < n) d[i] = rsqrtf(d[i]);
}

__global__ void k_zero(float* p, int n) {
    int i = blockIdx.x * blockDim.x + threadIdx.x;
    if (i < n) p[i] = 0.0f;
}

// ---------------- pre-pack W[KIN x 64] (f32, row-major) into f16 WMMA B-lane layout ----------
// Wp element order: [(kc*4 + nt)*32 + lane] -> v16h holding the 16 f16 values that lane
// needs for chunk kc, n-tile nt (per CDNA5 ISA 7.12.2 16-bit B layout).
template <int KIN>
__global__ void k_prepack_w(const float* __restrict__ W, _Float16* __restrict__ Wp) {
    int tid = blockIdx.x * blockDim.x + threadIdx.x;
    if (tid >= (KIN / 32) * 4 * 32) return;
    const int lane = tid & 31;
    const int nt   = (tid >> 5) & 3;
    const int kc   = tid >> 7;
    const int half = lane >> 4;
    const int l16  = lane & 15;
    const int kb1  = kc * 32 + half * 8;
    const int kb2  = kb1 + 16;
    const int n    = nt * 16 + l16;
    _Float16* o = Wp + (size_t)tid * 16;
#pragma unroll
    for (int j = 0; j < 8; ++j) o[j]     = (_Float16)W[(size_t)(kb1 + j) * HIDDEN + n];
#pragma unroll
    for (int j = 0; j < 8; ++j) o[8 + j] = (_Float16)W[(size_t)(kb2 + j) * HIDDEN + n];
}

// ---------------- WMMA GEMM: out[N x 64] = X[N x KIN] @ W[KIN x 64] ----------------
// One wave computes a 16-row x 64-col strip: 4 accumulators, A regs reused 4x.
// B tiles come pre-packed in f16 lane layout -> two global_load_b128 per tile.
template <int KIN>
__global__ __launch_bounds__(128) void k_gemm_wmma(const float* __restrict__ X,
                                                   const _Float16* __restrict__ Wp,
                                                   float* __restrict__ out,
                                                   int nrows) {
    const int lane = threadIdx.x & 31;
    const int wave = threadIdx.x >> 5;
    const int row0 = blockIdx.x * 64 + wave * 16;
    if (row0 >= nrows) return;

    const int half = lane >> 4;          // 0: K 0-7/16-23, 1: K 8-15/24-31
    const int l16  = lane & 15;
    const int r    = row0 + l16;         // A-matrix row owned by this lane

    const v16h* Wv = reinterpret_cast<const v16h*>(Wp);

    v8f acc[4] = {v8f{}, v8f{}, v8f{}, v8f{}};

    for (int k0 = 0; k0 < KIN; k0 += 32) {
        const int kc  = k0 >> 5;
        const int kb1 = k0 + half * 8;
        const int kb2 = kb1 + 16;
        const float* xr = X + (size_t)r * KIN;
        float4 p0 = *reinterpret_cast<const float4*>(xr + kb1);
        float4 p1 = *reinterpret_cast<const float4*>(xr + kb1 + 4);
        float4 p2 = *reinterpret_cast<const float4*>(xr + kb2);
        float4 p3 = *reinterpret_cast<const float4*>(xr + kb2 + 4);
        v16h a;
        a[0]=(_Float16)p0.x; a[1]=(_Float16)p0.y; a[2]=(_Float16)p0.z; a[3]=(_Float16)p0.w;
        a[4]=(_Float16)p1.x; a[5]=(_Float16)p1.y; a[6]=(_Float16)p1.z; a[7]=(_Float16)p1.w;
        a[8]=(_Float16)p2.x; a[9]=(_Float16)p2.y; a[10]=(_Float16)p2.z; a[11]=(_Float16)p2.w;
        a[12]=(_Float16)p3.x; a[13]=(_Float16)p3.y; a[14]=(_Float16)p3.z; a[15]=(_Float16)p3.w;

#pragma unroll
        for (int nt = 0; nt < 4; ++nt) {
            v16h b = Wv[(size_t)(kc * 4 + nt) * 32 + lane];
            acc[nt] = __builtin_amdgcn_wmma_f32_16x16x32_f16(
                false, a, false, b, (short)0, acc[nt], false, false);
        }
    }

    // C/D layout: VGPR v -> row (v + half*8), lane l16 -> column
#pragma unroll
    for (int nt = 0; nt < 4; ++nt) {
#pragma unroll
        for (int v = 0; v < 8; ++v) {
            int rr = row0 + v + half * 8;
            out[(size_t)rr * HIDDEN + nt * 16 + l16] = acc[nt][v];
        }
    }
}

// ---------------- edge message passing: agg[dst] += h[src] * dinv[src]*dinv[dst] ----------------
// One wave per edge (wave-uniform indices scalarized), 2 columns per lane.
__global__ __launch_bounds__(256) void k_edge_scatter(const float* __restrict__ h,
                                                      const long long* __restrict__ src,
                                                      const long long* __restrict__ dst,
                                                      const float* __restrict__ dinv,
                                                      float* __restrict__ agg, int n_edges) {
    const int lane = threadIdx.x & 31;
    const int e    = blockIdx.x * 8 + (threadIdx.x >> 5);   // 8 waves per block
    if (e >= n_edges) return;
    int s = (int)src[e];
    int d = (int)dst[e];
    s = __builtin_amdgcn_readfirstlane(s);                  // wave-uniform -> SGPR
    d = __builtin_amdgcn_readfirstlane(d);
    const float norm = dinv[s] * dinv[d];                   // scalar loads
    const int c = lane * 2;
    float2 hv = *reinterpret_cast<const float2*>(h + (size_t)s * HIDDEN + c);
    float* ap = agg + (size_t)d * HIDDEN + c;
    unsafeAtomicAdd(ap,     hv.x * norm);
    unsafeAtomicAdd(ap + 1, hv.y * norm);
}

// ---------------- agg = relu(agg + h * dinv^2 + b) ----------------
__global__ void k_selfloop_bias_relu(float* __restrict__ agg,
                                     const float* __restrict__ h,
                                     const float* __restrict__ dinv,
                                     const float* __restrict__ b, int n) {
    int idx = blockIdx.x * blockDim.x + threadIdx.x;
    if (idx >= n * HIDDEN) return;
    int i = idx >> 6;
    int c = idx & 63;
    float di = dinv[i];
    float v = agg[idx] + h[idx] * di * di + b[c];
    agg[idx] = fmaxf(v, 0.0f);
}

// ---------------- global mean pool (stage 1): LDS-accumulated segment sums ----------------
__global__ __launch_bounds__(256) void k_pool(const float* __restrict__ h,
                                              const long long* __restrict__ batch,
                                              float* __restrict__ sums,
                                              float* __restrict__ cnts, int n) {
    __shared__ float ls[N_GRAPHS * HIDDEN];   // 16 KB
    __shared__ float lc[N_GRAPHS];
    for (int i = threadIdx.x; i < N_GRAPHS * HIDDEN; i += 256) ls[i] = 0.0f;
    for (int i = threadIdx.x; i < N_GRAPHS;          i += 256) lc[i] = 0.0f;
    __syncthreads();

    const int c   = threadIdx.x & 63;
    const int sub = threadIdx.x >> 6;        // 0..3
    const int per = (n + gridDim.x - 1) / gridDim.x;
    const int start = blockIdx.x * per;
    const int end   = min(n, start + per);
    for (int i = start + sub; i < end; i += 4) {
        int g = (int)batch[i];
        atomicAdd(&ls[g * HIDDEN + c], h[(size_t)i * HIDDEN + c]);   // ds_add_f32
        if (c == 0) atomicAdd(&lc[g], 1.0f);
    }
    __syncthreads();

    for (int i = threadIdx.x; i < N_GRAPHS * HIDDEN; i += 256)
        if (ls[i] != 0.0f) unsafeAtomicAdd(&sums[i], ls[i]);
    for (int i = threadIdx.x; i < N_GRAPHS; i += 256)
        if (lc[i] != 0.0f) unsafeAtomicAdd(&cnts[i], lc[i]);
}

// ---------------- head: pooled @ Wlin + blin, log_softmax ----------------
__global__ void k_classify(const float* __restrict__ sums,
                           const float* __restrict__ cnts,
                           const float* __restrict__ Wlin,
                           const float* __restrict__ blin,
                           float* __restrict__ out) {
    int g = threadIdx.x;
    if (g >= N_GRAPHS) return;
    float inv = 1.0f / fmaxf(cnts[g], 1.0f);
    float logits[N_CLASSES];
#pragma unroll
    for (int j = 0; j < N_CLASSES; ++j) logits[j] = blin[j];
    for (int c = 0; c < HIDDEN; ++c) {
        float p = sums[g * HIDDEN + c] * inv;
#pragma unroll
        for (int j = 0; j < N_CLASSES; ++j) logits[j] += p * Wlin[c * N_CLASSES + j];
    }
    float m = logits[0];
#pragma unroll
    for (int j = 1; j < N_CLASSES; ++j) m = fmaxf(m, logits[j]);
    float s = 0.0f;
#pragma unroll
    for (int j = 0; j < N_CLASSES; ++j) s += expf(logits[j] - m);
    float lse = m + logf(s);
#pragma unroll
    for (int j = 0; j < N_CLASSES; ++j) out[g * N_CLASSES + j] = logits[j] - lse;
}

extern "C" void kernel_launch(void* const* d_in, const int* in_sizes, int n_in,
                              void* d_out, int out_size, void* d_ws, size_t ws_size,
                              hipStream_t stream) {
    const float*     x    = (const float*)d_in[0];
    const long long* ei   = (const long long*)d_in[1];   // [2, E] int64
    const long long* batch= (const long long*)d_in[2];   // [N] int64
    const float*     W1   = (const float*)d_in[3];
    const float*     b1   = (const float*)d_in[4];
    const float*     W2   = (const float*)d_in[5];
    const float*     b2   = (const float*)d_in[6];
    const float*     Wlin = (const float*)d_in[7];
    const float*     blin = (const float*)d_in[8];
    float*           out  = (float*)d_out;

    const int N = N_NODES, E = N_EDGES;
    const long long* src = ei;
    const long long* dst = ei + E;

    // workspace layout (all offsets >=32B-aligned)
    float*     dinv = (float*)d_ws;                    // N
    float*     bufA = dinv + N;                        // N*HIDDEN  (h)
    float*     bufB = bufA + (size_t)N * HIDDEN;       // N*HIDDEN  (agg)
    float*     sums = bufB + (size_t)N * HIDDEN;       // 64*64
    float*     cnts = sums + N_GRAPHS * HIDDEN;        // 64
    _Float16*  Wp1  = (_Float16*)(cnts + N_GRAPHS);    // 128*64 f16 (packed)
    _Float16*  Wp2  = Wp1 + (size_t)N_FEAT * HIDDEN;   // 64*64 f16 (packed)

    const int T = 256;
    const int gemm_grid = (N + 63) / 64;

    // normalization + weight packing (independent, all on stream)
    k_init_deg<<<(N + T - 1) / T, T, 0, stream>>>(dinv, N);
    k_count_deg<<<(E + T - 1) / T, T, 0, stream>>>(dst, dinv, E);
    k_finalize_dinv<<<(N + T - 1) / T, T, 0, stream>>>(dinv, N);
    k_prepack_w<N_FEAT><<<4, 128, 0, stream>>>(W1, Wp1);   // 512 threads
    k_prepack_w<HIDDEN><<<2, 128, 0, stream>>>(W2, Wp2);   // 256 threads

    // ---- layer 1 ----
    k_gemm_wmma<N_FEAT><<<gemm_grid, 128, 0, stream>>>(x, Wp1, bufA, N);
    k_zero<<<((N * HIDDEN) + T - 1) / T, T, 0, stream>>>(bufB, N * HIDDEN);
    k_edge_scatter<<<(E + 7) / 8, 256, 0, stream>>>(bufA, src, dst, dinv, bufB, E);
    k_selfloop_bias_relu<<<((N * HIDDEN) + T - 1) / T, T, 0, stream>>>(bufB, bufA, dinv, b1, N);

    // ---- layer 2 ----
    k_gemm_wmma<HIDDEN><<<gemm_grid, 128, 0, stream>>>(bufB, Wp2, bufA, N);
    k_zero<<<((N * HIDDEN) + T - 1) / T, T, 0, stream>>>(bufB, N * HIDDEN);
    k_edge_scatter<<<(E + 7) / 8, 256, 0, stream>>>(bufA, src, dst, dinv, bufB, E);
    k_selfloop_bias_relu<<<((N * HIDDEN) + T - 1) / T, T, 0, stream>>>(bufB, bufA, dinv, b2, N);

    // ---- pool + head ----
    k_zero<<<(N_GRAPHS * HIDDEN + N_GRAPHS + T - 1) / T, T, 0, stream>>>(sums, N_GRAPHS * HIDDEN + N_GRAPHS);
    k_pool<<<128, 256, 0, stream>>>(bufB, batch, sums, cnts, N);
    k_classify<<<1, 64, 0, stream>>>(sums, cnts, Wlin, blin, out);
}
